// AttentionWithRoPE_5858335391788
// MI455X (gfx1250) — matrix operand order repaired and checked by reading, not compiled, not run on hardware
//
#include <hip/hip_runtime.h>
#include <hip/hip_bf16.h>

#define BB   2
#define LL   2048
#define DD   1024
#define HH   16
#define HDIM 64
#define MM   (BB * LL)

typedef __attribute__((ext_vector_type(16))) __bf16 v16bf;
typedef __attribute__((ext_vector_type(8)))  float  v8f;
typedef __attribute__((ext_vector_type(4)))  int    v4i;

#if __has_builtin(__builtin_amdgcn_global_load_async_to_lds_b128)
#define HAVE_ASYNC_LDS 1
#else
#define HAVE_ASYNC_LDS 0
#endif

// pointer-to-(v4i in global AS) and pointer-to-(v4i in LDS AS)
typedef __attribute__((address_space(1))) v4i* gbl_v4i;
typedef __attribute__((address_space(3))) v4i* lds_v4i;

union Frag { v16bf v; unsigned int u[8]; };

// ---- async 16-byte global->LDS copy (ASYNCcnt-tracked on gfx1250) ----------
__device__ __forceinline__ void async_b128(__bf16* ldst, const __bf16* gsrc) {
#if HAVE_ASYNC_LDS
  __builtin_amdgcn_global_load_async_to_lds_b128((gbl_v4i)gsrc, (lds_v4i)ldst,
                                                 0, 0);
#else
  *(uint4*)ldst = *(const uint4*)gsrc;
#endif
}

__device__ __forceinline__ void wait_async0() {
#if HAVE_ASYNC_LDS && __has_builtin(__builtin_amdgcn_s_wait_asynccnt)
  __builtin_amdgcn_s_wait_asynccnt(0);
#else
  asm volatile("s_wait_asynccnt 0x0" ::: "memory");
#endif
}

// ---- CDNA5 WMMA fragment loaders (per ISA 7.12.2 VGPR layouts) -------------
__device__ __forceinline__ v16bf ld_frag_a(const __bf16* base, int pitch,
                                           int mbase, int kpair_off) {
  const int lane = threadIdx.x & 31;
  const int m  = mbase + (lane & 15);
  const int hf = lane >> 4;
  const unsigned int* rp = (const unsigned int*)(base + (size_t)m * pitch) + kpair_off;
  Frag f;
#pragma unroll
  for (int i = 0; i < 4; ++i) f.u[i]     = rp[hf * 4 + i];
#pragma unroll
  for (int i = 0; i < 4; ++i) f.u[4 + i] = rp[8 + hf * 4 + i];
  return f.v;
}

__device__ __forceinline__ v16bf ld_frag_b(const __bf16* base, int pitch,
                                           int nbase, int kpair_off) {
  const int lane = threadIdx.x & 31;
  const int n  = nbase + (lane & 15);
  const int kh = lane >> 4;
  const unsigned int* rp = (const unsigned int*)(base + (size_t)n * pitch) + kpair_off;
  Frag f;
#pragma unroll
  for (int i = 0; i < 4; ++i) f.u[i]     = rp[kh * 4 + i];
#pragma unroll
  for (int i = 0; i < 4; ++i) f.u[4 + i] = rp[8 + kh * 4 + i];
  return f.v;
}

__device__ __forceinline__ v8f wmma_bf16(v16bf a, v16bf b, v8f c) {
  return __builtin_amdgcn_wmma_f32_16x16x32_bf16(false, a, false, b,
                                                 (short)0, c, false, false);
}

__device__ __forceinline__ v8f v8f_zero() {
  v8f z;
#pragma unroll
  for (int e = 0; e < 8; ++e) z[e] = 0.0f;
  return z;
}

// ---- K0: fp32 -> bf16 conversion -------------------------------------------
__global__ __launch_bounds__(256) void cvt_bf16(const float* __restrict__ s,
                                                __bf16* __restrict__ d, int n4) {
  int i = blockIdx.x * 256 + threadIdx.x;
  if (i < n4) {
    float4 x = ((const float4*)s)[i];
    union { __bf16 h[4]; uint2 u; } t;
    t.h[0] = (__bf16)x.x; t.h[1] = (__bf16)x.y;
    t.h[2] = (__bf16)x.z; t.h[3] = (__bf16)x.w;
    ((uint2*)d)[i] = t.u;
  }
}

// ---- K1: QKV projection (async double-buffered bf16 WMMA) + RoPE -----------
// grid = (N/128=8, M/128=32, 3), block = 256 (8 waves, each 32x64)
__global__ __launch_bounds__(256) void qkv_proj_rope(
    const __bf16* __restrict__ qx, const __bf16* __restrict__ kx,
    const __bf16* __restrict__ vx,
    const __bf16* __restrict__ wq, const __bf16* __restrict__ wk,
    const __bf16* __restrict__ wv,
    const float* __restrict__ bq, const float* __restrict__ bk,
    const float* __restrict__ bv,
    __bf16* __restrict__ qp, __bf16* __restrict__ kp, __bf16* __restrict__ vT) {
  __shared__ __bf16 As[2][128 * 40];
  __shared__ __bf16 Bs[2][128 * 40];

  const int z = blockIdx.z;
  const __bf16* X    = (z == 0) ? qx : (z == 1) ? kx : vx;
  const __bf16* W    = (z == 0) ? wq : (z == 1) ? wk : wv;
  const float*  bias = (z == 0) ? bq : (z == 1) ? bk : bv;

  const int mBase = blockIdx.y * 128;
  const int nBase = blockIdx.x * 128;
  const int tid = threadIdx.x, wid = tid >> 5;
  const int wm = wid & 3, wn = wid >> 2;
  const int lane = tid & 31, laneM = lane & 15, laneH = lane >> 4;

  auto stage = [&](int k0, int buf) {
#pragma unroll
    for (int i = 0; i < 2; ++i) {
      int e = tid + i * 256;             // 0..511
      int row = e >> 2, c = (e & 3) << 3;
      async_b128(&As[buf][row * 40 + c],
                 X + (size_t)(mBase + row) * DD + k0 + c);
      async_b128(&Bs[buf][row * 40 + c],
                 W + (size_t)(nBase + row) * DD + k0 + c);
    }
  };

  v8f acc[2][4];
#pragma unroll
  for (int i = 0; i < 2; ++i)
#pragma unroll
    for (int j = 0; j < 4; ++j) acc[i][j] = v8f_zero();

  stage(0, 0);
  const int NK = DD / 32;
  for (int kb = 0; kb < NK; ++kb) {
    wait_async0();       // my stage-kb copies landed
    __syncthreads();     // everyone's landed; prev compute done (WAR safe)
    if (kb + 1 < NK) stage((kb + 1) * 32, (kb + 1) & 1);  // overlaps WMMA

    const __bf16* as = As[kb & 1];
    const __bf16* bs = Bs[kb & 1];
    v16bf a0 = ld_frag_a(as, 40, wm * 32, 0);
    v16bf a1 = ld_frag_a(as, 40, wm * 32 + 16, 0);
#pragma unroll
    for (int t = 0; t < 4; ++t) {
      v16bf bfr = ld_frag_b(bs, 40, wn * 64 + t * 16, 0);
      acc[0][t] = wmma_bf16(a0, bfr, acc[0][t]);
      acc[1][t] = wmma_bf16(a1, bfr, acc[1][t]);
    }
  }

  const int colBase = nBase + wn * 64;   // 64-aligned: wave spans one head
  const float oscale = (z == 0) ? 0.125f : 1.0f;  // fold 1/sqrt(HD) into Q
  float invf[2];
#pragma unroll
  for (int t = 0; t < 2; ++t) {
    int d = (colBase + t * 16 + laneM) & 63;     // 0..31
    invf[t] = __powf(10000.0f, -(float)d * (1.0f / 32.0f));
  }

#pragma unroll
  for (int mt = 0; mt < 2; ++mt) {
#pragma unroll
    for (int r = 0; r < 8; ++r) {
      int row = mBase + wm * 32 + mt * 16 + laneH * 8 + r;
      int b = row >> 11, l = row & (LL - 1);
      if (z < 2) {
        __bf16* dst = (z == 0) ? qp : kp;
#pragma unroll
        for (int t = 0; t < 2; ++t) {    // pair (d, d+32) = tiles (t, t+2)
          int col = colBase + t * 16 + laneM;
          int d = col & 63;
          int h = col >> 6;
          float x1 = acc[mt][t][r]     + bias[col];
          float x2 = acc[mt][t + 2][r] + bias[col + 32];
          float sn, cs;
          __sincosf((float)l * invf[t], &sn, &cs);
          __bf16* o = dst + ((size_t)(b * HH + h) * LL + l) * HDIM;
          o[d]      = (__bf16)((x1 * cs - x2 * sn) * oscale);
          o[d + 32] = (__bf16)((x2 * cs + x1 * sn) * oscale);
        }
      } else {  // V: store transposed per head -> vT[b][h][hd][l]
#pragma unroll
        for (int t = 0; t < 4; ++t) {
          int col = colBase + t * 16 + laneM;
          int hd = col & 63, h = col >> 6;
          vT[((size_t)(b * HH + h) * HDIM + hd) * LL + l] =
              (__bf16)(acc[mt][t][r] + bias[col]);
        }
      }
    }
  }
}

// ---- K2: flash attention per (b,h); grid = (L/128=16, H=16, B=2) -----------
__global__ __launch_bounds__(256) void attn_kernel(
    const __bf16* __restrict__ qp, const __bf16* __restrict__ kp,
    const __bf16* __restrict__ vT, __bf16* __restrict__ xo) {
  __shared__ __bf16 Kt[2][64 * 72];   // [kv][hd]  (B-frags for Q.K^T)
  __shared__ __bf16 Vt[2][64 * 72];   // [hd][kv]  (B-frags for P.V)
  __shared__ __bf16 Pw[8][16 * 72];   // per-wave P staging

  const int h = blockIdx.y, b = blockIdx.z;
  const int tid = threadIdx.x, wid = tid >> 5;
  const int lane = tid & 31, laneM = lane & 15, laneH = lane >> 4;
  const int q0 = blockIdx.x * 128 + wid * 16;

  const size_t headQK = (size_t)(b * HH + h) * LL * HDIM;
  const size_t headV  = (size_t)(b * HH + h) * HDIM * LL;

  auto stageKV = [&](int kv0, int buf) {
#pragma unroll
    for (int i = 0; i < 2; ++i) {
      int e = tid + i * 256;
      int row = e >> 3, c = (e & 7) << 3;
      async_b128(&Kt[buf][row * 72 + c],
                 kp + headQK + (size_t)(kv0 + row) * HDIM + c);
      async_b128(&Vt[buf][row * 72 + c],
                 vT + headV + (size_t)row * LL + kv0 + c);
    }
  };

  // Q A-fragments straight from global (hd-contiguous rows; pre-scaled)
  const unsigned int* qrow =
      (const unsigned int*)(qp + headQK + (size_t)(q0 + laneM) * HDIM);
  Frag qa0, qa1;
#pragma unroll
  for (int i = 0; i < 4; ++i) {
    qa0.u[i]     = qrow[laneH * 4 + i];
    qa0.u[4 + i] = qrow[8 + laneH * 4 + i];
    qa1.u[i]     = qrow[16 + laneH * 4 + i];
    qa1.u[4 + i] = qrow[24 + laneH * 4 + i];
  }

  float mrow[8], lrow[8];
  v8f o[4];
#pragma unroll
  for (int r = 0; r < 8; ++r) { mrow[r] = -1e30f; lrow[r] = 0.0f; }
#pragma unroll
  for (int t = 0; t < 4; ++t) o[t] = v8f_zero();

  stageKV(0, 0);
  const int NKV = LL / 64;
  for (int kvb = 0; kvb < NKV; ++kvb) {
    wait_async0();
    __syncthreads();
    if (kvb + 1 < NKV) stageKV((kvb + 1) * 64, (kvb + 1) & 1);

    const __bf16* kt = Kt[kvb & 1];
    const __bf16* vt = Vt[kvb & 1];

    // S = Q.K^T (4 kv-tiles x 2 k-steps); scale pre-folded into Q
    v8f s[4];
#pragma unroll
    for (int t = 0; t < 4; ++t) {
      s[t] = v8f_zero();
      v16bf b0 = ld_frag_b(kt, 72, t * 16, 0);
      s[t] = wmma_bf16(qa0.v, b0, s[t]);
      v16bf b1 = ld_frag_b(kt, 72, t * 16, 16);
      s[t] = wmma_bf16(qa1.v, b1, s[t]);
    }

    // online softmax: rows 0..7 in lanes 0-15, rows 8..15 in lanes 16-31
    float p[4][8];
#pragma unroll
    for (int r = 0; r < 8; ++r) {
      float vmax = s[0][r];
#pragma unroll
      for (int t = 1; t < 4; ++t) vmax = fmaxf(vmax, s[t][r]);
#pragma unroll
      for (int off = 1; off < 16; off <<= 1)
        vmax = fmaxf(vmax, __shfl_xor(vmax, off, 32));
      float mnew = fmaxf(mrow[r], vmax);
      float corr = __expf(mrow[r] - mnew);
      float rs = 0.0f;
#pragma unroll
      for (int t = 0; t < 4; ++t) {
        float pe = __expf(s[t][r] - mnew);
        p[t][r] = pe;
        rs += pe;
      }
#pragma unroll
      for (int off = 1; off < 16; off <<= 1) rs += __shfl_xor(rs, off, 32);
      lrow[r] = lrow[r] * corr + rs;
      mrow[r] = mnew;
#pragma unroll
      for (int t = 0; t < 4; ++t) o[t][r] *= corr;
    }

    // D-layout P -> per-wave LDS tile -> A-layout fragments
    __bf16* pw = Pw[wid];
#pragma unroll
    for (int t = 0; t < 4; ++t)
#pragma unroll
      for (int r = 0; r < 8; ++r)
        pw[(laneH * 8 + r) * 72 + t * 16 + laneM] = (__bf16)p[t][r];
    asm volatile("s_wait_dscnt 0x0" ::: "memory");

    v16bf pa0 = ld_frag_a(pw, 72, 0, 0);
    v16bf pa1 = ld_frag_a(pw, 72, 0, 16);
#pragma unroll
    for (int t = 0; t < 4; ++t) {
      v16bf vb0 = ld_frag_b(vt, 72, t * 16, 0);
      o[t] = wmma_bf16(pa0, vb0, o[t]);
      v16bf vb1 = ld_frag_b(vt, 72, t * 16, 16);
      o[t] = wmma_bf16(pa1, vb1, o[t]);
    }
  }

  // normalize + write attn output in [B*L][DIM] bf16 for the final GEMM
#pragma unroll
  for (int r = 0; r < 8; ++r) {
    float invl = 1.0f / lrow[r];
    int row = q0 + laneH * 8 + r;
#pragma unroll
    for (int t = 0; t < 4; ++t)
      xo[(size_t)(b * LL + row) * DD + h * HDIM + t * 16 + laneM] =
          (__bf16)(o[t][r] * invl);
  }
}

// ---- K3: output projection (async double-buffered), fp32 epilogue + bias ---
__global__ __launch_bounds__(256) void out_proj(
    const __bf16* __restrict__ X, const __bf16* __restrict__ W,
    const float* __restrict__ bias, float* __restrict__ out) {
  __shared__ __bf16 As[2][128 * 40];
  __shared__ __bf16 Bs[2][128 * 40];
  const int mBase = blockIdx.y * 128;
  const int nBase = blockIdx.x * 128;
  const int tid = threadIdx.x, wid = tid >> 5;
  const int wm = wid & 3, wn = wid >> 2;
  const int lane = tid & 31, laneM = lane & 15, laneH = lane >> 4;

  auto stage = [&](int k0, int buf) {
#pragma unroll
    for (int i = 0; i < 2; ++i) {
      int e = tid + i * 256;
      int row = e >> 2, c = (e & 3) << 3;
      async_b128(&As[buf][row * 40 + c],
                 X + (size_t)(mBase + row) * DD + k0 + c);
      async_b128(&Bs[buf][row * 40 + c],
                 W + (size_t)(nBase + row) * DD + k0 + c);
    }
  };

  v8f acc[2][4];
#pragma unroll
  for (int i = 0; i < 2; ++i)
#pragma unroll
    for (int j = 0; j < 4; ++j) acc[i][j] = v8f_zero();

  stage(0, 0);
  const int NK = DD / 32;
  for (int kb = 0; kb < NK; ++kb) {
    wait_async0();
    __syncthreads();
    if (kb + 1 < NK) stage((kb + 1) * 32, (kb + 1) & 1);

    const __bf16* as = As[kb & 1];
    const __bf16* bs = Bs[kb & 1];
    v16bf a0 = ld_frag_a(as, 40, wm * 32, 0);
    v16bf a1 = ld_frag_a(as, 40, wm * 32 + 16, 0);
#pragma unroll
    for (int t = 0; t < 4; ++t) {
      v16bf bfr = ld_frag_b(bs, 40, wn * 64 + t * 16, 0);
      acc[0][t] = wmma_bf16(a0, bfr, acc[0][t]);
      acc[1][t] = wmma_bf16(a1, bfr, acc[1][t]);
    }
  }
#pragma unroll
  for (int mt = 0; mt < 2; ++mt)
#pragma unroll
    for (int r = 0; r < 8; ++r) {
      int row = mBase + wm * 32 + mt * 16 + laneH * 8 + r;
#pragma unroll
      for (int t = 0; t < 4; ++t) {
        int col = nBase + wn * 64 + t * 16 + laneM;
        out[(size_t)row * DD + col] = acc[mt][t][r] + bias[col];
      }
    }
}

// ---- host ------------------------------------------------------------------
extern "C" void kernel_launch(void* const* d_in, const int* in_sizes, int n_in,
                              void* d_out, int out_size, void* d_ws,
                              size_t ws_size, hipStream_t stream) {
  (void)in_sizes; (void)n_in; (void)out_size; (void)ws_size;
  const float* q  = (const float*)d_in[0];
  const float* k  = (const float*)d_in[1];
  const float* v  = (const float*)d_in[2];
  const float* Wq = (const float*)d_in[3];
  const float* bq = (const float*)d_in[4];
  const float* Wk = (const float*)d_in[5];
  const float* bk = (const float*)d_in[6];
  const float* Wv = (const float*)d_in[7];
  const float* bv = (const float*)d_in[8];
  const float* Wo = (const float*)d_in[9];
  const float* bo = (const float*)d_in[10];
  float* out = (float*)d_out;

  char* ws = (char*)d_ws;
  const size_t WSZ = (size_t)DD * DD * sizeof(__bf16);   // 2 MB
  const size_t PSZ = (size_t)MM * DD * sizeof(__bf16);   // 8 MB
  size_t off = 0;
  __bf16* wq16 = (__bf16*)(ws + off); off += WSZ;
  __bf16* wk16 = (__bf16*)(ws + off); off += WSZ;
  __bf16* wv16 = (__bf16*)(ws + off); off += WSZ;
  __bf16* wo16 = (__bf16*)(ws + off); off += WSZ;
  __bf16* qx   = (__bf16*)(ws + off); off += PSZ;
  __bf16* kx   = (__bf16*)(ws + off); off += PSZ;
  __bf16* vx   = (__bf16*)(ws + off); off += PSZ;
  __bf16* qp   = (__bf16*)(ws + off); off += PSZ;
  __bf16* kp   = (__bf16*)(ws + off); off += PSZ;
  __bf16* vT   = (__bf16*)(ws + off); off += PSZ;
  __bf16* xo   = (__bf16*)(ws + off); off += PSZ;

  const int w4 = DD * DD / 4;
  cvt_bf16<<<w4 / 256, 256, 0, stream>>>(Wq, wq16, w4);
  cvt_bf16<<<w4 / 256, 256, 0, stream>>>(Wk, wk16, w4);
  cvt_bf16<<<w4 / 256, 256, 0, stream>>>(Wv, wv16, w4);
  cvt_bf16<<<w4 / 256, 256, 0, stream>>>(Wo, wo16, w4);
  const int x4 = MM * DD / 4;
  cvt_bf16<<<x4 / 256, 256, 0, stream>>>(q, qx, x4);
  cvt_bf16<<<x4 / 256, 256, 0, stream>>>(k, kx, x4);
  cvt_bf16<<<x4 / 256, 256, 0, stream>>>(v, vx, x4);

  dim3 g1(DD / 128, MM / 128, 3);
  qkv_proj_rope<<<g1, 256, 0, stream>>>(qx, kx, vx, wq16, wk16, wv16,
                                        bq, bk, bv, qp, kp, vT);

  dim3 g2(LL / 128, HH, BB);
  attn_kernel<<<g2, 256, 0, stream>>>(qp, kp, vT, xo);

  dim3 g3(DD / 128, MM / 128, 1);
  out_proj<<<g3, 256, 0, stream>>>(xo, wo16, bo, out);
}
